// IFRM2Mv1_39462159515693
// MI455X (gfx1250) — compile-verified, matrix operand order checked by reading, not run on hardware
//
#include <hip/hip_runtime.h>
#include <hip/hip_bf16.h>
#include <stddef.h>

// =====================================================================
// IFRNet-style frame interpolation for MI455X (gfx1250, wave32, WMMA).
//
// Core engine: implicit-GEMM 3x3 conv using v_wmma_f32_16x16x32_bf16.
//   - M = Cout, N = output pixels (128-wide row tiles).
//   - Virtual K axis is tap-major: k = tap*Cin32 + c (Cin32 = Cin padded
//     to 32), so a 32-K chunk never straddles a tap -> tap/iy/row pointer
//     are uniform per chunk and per-thread validity is ONE bool.
//   - Branchless clamped loads -> v_cvt_pk_bf16_f32 -> one ds_store_b128
//     per octet in WMMA *fragment order*; each lane reads its fragment as
//     one aligned 32B LDS load.
//   - 256 threads = 8 waves; tile 64x128; each wave: 1 A-frag x 4 B-frags
//     = 4 WMMA per chunk; software-pipelined global loads.
// Deconv stages its weight slice into LDS via the Tensor Data Mover
// (tensor_load_to_lds + s_wait_tensorcnt), 6-arg clang-23 builtin form.
//
// Param leaves assumed in jax tree_leaves order (sorted dict keys):
//   0: alpha
//   1..20  : dec1  (cr{a,b,w}, rb{a, c1{a,b,w}..c4{a,b,w}, c5{b,w}}, up{b,w})
//   21..40 : dec2,  41..60 : dec3,  61..80 : dec4
//   81..104: enc (8 x convrelu {a,b,w})
//   105: x0, 106: x1, 107: xt, 108: t
// =====================================================================

typedef __attribute__((ext_vector_type(16))) __bf16 v16bf;
typedef __attribute__((ext_vector_type(8)))  float  v8f;
typedef __attribute__((ext_vector_type(4)))  unsigned v4u;
typedef __attribute__((ext_vector_type(8)))  int    v8i;
typedef __attribute__((ext_vector_type(4)))  int    v4i;
typedef __attribute__((address_space(3)))    float  lds_float;

static_assert(sizeof(__hip_bfloat162) == 4, "bf16x2 must be 4 bytes");

__device__ __forceinline__ unsigned pkbf(float a, float b) {
  __hip_bfloat162 h = __float22bfloat162_rn(make_float2(a, b));  // v_cvt_pk_bf16_f32
  unsigned u;
  __builtin_memcpy(&u, &h, sizeof(u));
  return u;
}

// ---------------------------------------------------------------------
// Implicit-GEMM 3x3 conv (pad=1, stride 1 or 2) + bias + optional
// residual-add + optional PReLU.  bf16 WMMA, f32 accumulate.
// ---------------------------------------------------------------------
__launch_bounds__(256)
__global__ void k_conv3x3_wmma(
    const float* __restrict__ in, int CinTot, int cin_off, int Cin,
    int H, int W,
    float* __restrict__ out, int CoutTot, int cout_off, int Cout,
    int Ho, int Wo, int stride,
    const float* __restrict__ wgt, const float* __restrict__ bias,
    const float* __restrict__ prelu_a,
    const float* __restrict__ res, int resCT, int res_off)
{
  __shared__ alignas(64) unsigned short lsA[64 * 32];   // weights  (MxK)
  __shared__ alignas(64) unsigned short lsB[32 * 128];  // im2col   (KxN)

  const int tid  = threadIdx.x;
  const int lane = tid & 31;
  const int wv   = tid >> 5;
  const int mi   = wv & 3;          // wave's M block (0..3)
  const int nh   = wv >> 2;         // wave's N quad  (0..1) -> 4 n-blocks

  const int n0 = blockIdx.x * 128;
  const int oy = blockIdx.y % Ho;
  const int b  = blockIdx.y / Ho;
  const int m0 = blockIdx.z * 64;
  const int Kdim  = Cin * 9;
  const int Cin32 = (Cin + 31) & ~31;
  const size_t HWin = (size_t)H * W;

  v8f acc[4];
#pragma unroll
  for (int j = 0; j < 4; ++j) acc[j] = (v8f){0.f,0.f,0.f,0.f,0.f,0.f,0.f,0.f};

  // staging thread assignments
  const int aM = tid >> 2, aG = tid & 3;              // A: M row, K-octet
  const int am_cl = min(m0 + aM, Cout - 1);
  const bool am_ok = (m0 + aM) < Cout;
  const int bN = tid & 127, bG0 = (tid >> 7) & 1;     // B: N col, K-octet base
  const int ox_s = min(n0 + bN, Wo - 1) * stride;
  const int oy_s = oy * stride;

  float ra[8], rb[2][8];
  unsigned mA = 0;
  bool okB = false;

  auto loadA = [&](int tap_, int cc_) {
    const float* wrow = wgt + (size_t)am_cl * Kdim + tap_;
    int cb = cc_ + aG * 8;
    mA = 0;
#pragma unroll
    for (int j = 0; j < 8; ++j) {
      int c = cb + j;
      if (am_ok && c < Cin) mA |= (1u << j);
      ra[j] = wrow[(size_t)min(c, Cin - 1) * 9];
    }
  };
  auto loadB = [&](int cc_, int dy_, int dx_) {
    int iy = oy_s + dy_;
    int ix = ox_s + dx_;
    okB = ((unsigned)iy < (unsigned)H) && ((unsigned)ix < (unsigned)W);
    const float* prow = in + (((size_t)b * CinTot + cin_off) * H
                        + min(max(iy, 0), H - 1)) * W + min(max(ix, 0), W - 1);
    if (cc_ + 32 <= Cin) {                       // uniform fast path: no clamp
#pragma unroll
      for (int s = 0; s < 2; ++s) {
        int cb = cc_ + (bG0 | (s << 1)) * 8;
#pragma unroll
        for (int j = 0; j < 8; ++j) rb[s][j] = prow[(size_t)(cb + j) * HWin];
      }
    } else {
#pragma unroll
      for (int s = 0; s < 2; ++s) {
        int cb = cc_ + (bG0 | (s << 1)) * 8;
#pragma unroll
        for (int j = 0; j < 8; ++j)
          rb[s][j] = prow[(size_t)min(cb + j, Cin - 1) * HWin];
      }
    }
  };
  auto packA = [&]() {
    unsigned w[4];
#pragma unroll
    for (int wi = 0; wi < 4; ++wi) {
      float f0 = ((mA >> (2 * wi))     & 1u) ? ra[2 * wi]     : 0.f;
      float f1 = ((mA >> (2 * wi + 1)) & 1u) ? ra[2 * wi + 1] : 0.f;
      w[wi] = pkbf(f0, f1);
    }
    int L = (aM & 15) | ((aG & 1) << 4);
    *(uint4*)((char*)lsA + (aM >> 4) * 1024 + L * 32 + (aG >> 1) * 16) =
        (uint4){w[0], w[1], w[2], w[3]};
  };
  auto packB = [&]() {
#pragma unroll
    for (int s = 0; s < 2; ++s) {
      int g = bG0 | (s << 1);
      unsigned w[4];
#pragma unroll
      for (int wi = 0; wi < 4; ++wi)
        w[wi] = pkbf(okB ? rb[s][2 * wi] : 0.f, okB ? rb[s][2 * wi + 1] : 0.f);
      int L = (bN & 15) | ((g & 1) << 4);
      *(uint4*)((char*)lsB + (bN >> 4) * 1024 + L * 32 + (g >> 1) * 16) =
          (uint4){w[0], w[1], w[2], w[3]};
    }
  };

  int tap = 0, cc = 0, dy = -1, dx = -1;
  loadA(0, 0);
  loadB(0, -1, -1);
  const int NC = 9 * (Cin32 >> 5);
  for (int q = 0; q < NC; ++q) {
    packA();
    packB();
    __syncthreads();
    int cc_n = cc + 32, tap_n = tap, dy_n = dy, dx_n = dx;
    if (cc_n >= Cin32) {
      cc_n = 0; ++tap_n;
      if (++dx_n > 1) { dx_n = -1; ++dy_n; }
    }
    if (q + 1 < NC) { loadA(tap_n, cc_n); loadB(cc_n, dy_n, dx_n); }  // pipeline
    v16bf av = *(const v16bf*)((const char*)lsA + mi * 1024 + lane * 32);
#pragma unroll
    for (int j = 0; j < 4; ++j) {
      v16bf bv = *(const v16bf*)((const char*)lsB + (nh * 4 + j) * 1024 + lane * 32);
      acc[j] = __builtin_amdgcn_wmma_f32_16x16x32_bf16(false, av, false, bv,
                                                       (short)0, acc[j],
                                                       false, false);
    }
    __syncthreads();
    cc = cc_n; tap = tap_n; dy = dy_n; dx = dx_n;
  }

  // ---- epilogue: bias + residual + PReLU, masked store ----
#pragma unroll
  for (int j = 0; j < 4; ++j) {
    int nb = nh * 4 + j;
#pragma unroll
    for (int g = 0; g < 8; ++g) {
      int m  = mi * 16 + ((lane >> 4) << 3) + g;   // C/D layout: VGPR g -> M
      int nl = nb * 16 + (lane & 15);              // lane&15 -> N
      int co = m0 + m, ox = n0 + nl;
      if (co < Cout && ox < Wo) {
        float v = acc[j][g] + bias[co];
        if (res)
          v += res[(((size_t)b * resCT + res_off + co) * Ho + oy) * Wo + ox];
        if (prelu_a) { float al = prelu_a[co]; v = (v >= 0.f) ? v : al * v; }
        out[(((size_t)b * CoutTot + cout_off + co) * Ho + oy) * Wo + ox] = v;
      }
    }
  }
}

// ---------------------------------------------------------------------
// Transposed conv k=4 s=2 (lhs_dilation=2, pad=2): subpixel-parity form,
// 2x2 taps per output pixel.  Weight slice for this output channel is
// DMA'd into LDS by the Tensor Data Mover (D# per ISA ch.8, 2D tile of
// Cin*16 f32 elements), tracked with TENSORcnt.
// ---------------------------------------------------------------------
__global__ void k_deconv4x4s2(const float* __restrict__ in, int Cin, int H, int W,
                              float* __restrict__ out, int CoutTot, int Cout,
                              const float* __restrict__ wgt,
                              const float* __restrict__ bias)
{
  __shared__ alignas(16) float lsW[192 * 16];
  int zc = blockIdx.z;
  int b = zc / Cout, co = zc - b * Cout;

  if (threadIdx.x < 32) {          // one wave issues the TDM descriptor
    unsigned ldsoff = (unsigned)(unsigned long long)(lds_float*)lsW;
    unsigned long long ga =
        (unsigned long long)(const void*)(wgt + (size_t)co * Cin * 16);
    unsigned nel = (unsigned)(Cin * 16);            // elements (4B each)
    v4u g0;                     // D# group 0
    g0[0] = 1u;                                     // count=1, user mode
    g0[1] = ldsoff;                                 // lds_addr
    g0[2] = (unsigned)ga;                           // global_addr[31:0]
    g0[3] = (unsigned)((ga >> 32) & 0x01FFFFFFu) | 0x80000000u;  // type=2
    v8i g1;                     // D# group 1
    g1[0] = (int)(2u << 16);                        // data_size=4B
    g1[1] = (int)((nel & 0xFFFFu) << 16);           // tensor_dim0 lo
    g1[2] = (int)((nel >> 16) | (1u << 16));        // tensor_dim0 hi | dim1=1
    g1[3] = (int)((nel & 0xFFFFu) << 16);           // tile_dim0
    g1[4] = 1;                                      // tile_dim1 = 1
    g1[5] = (int)nel;                               // tensor_dim0_stride lo
    g1[6] = 0; g1[7] = 0;
    v4i z4 = {0, 0, 0, 0};
    v8i z8 = {0, 0, 0, 0, 0, 0, 0, 0};
    __builtin_amdgcn_tensor_load_to_lds(g0, g1, z4, z4, z8, 0);
    __builtin_amdgcn_s_wait_tensorcnt(0);
  }
  __syncthreads();

  int ox = blockIdx.x * blockDim.x + threadIdx.x;
  int oy = blockIdx.y;
  int Wo = 2 * W;
  if (ox >= Wo) return;
  int py = oy & 1, px = ox & 1;
  float s = bias[co];
  for (int ky = py; ky < 4; ky += 2) {
    int iy = (oy - 2 + ky) >> 1;
    if ((unsigned)iy >= (unsigned)H) continue;
    for (int kx = px; kx < 4; kx += 2) {
      int ix = (ox - 2 + kx) >> 1;
      if ((unsigned)ix >= (unsigned)W) continue;
      const float* ip = in + (((size_t)b * Cin) * H + iy) * W + ix;
      const float* wp = lsW + ky * 4 + kx;
      for (int ci = 0; ci < Cin; ++ci)
        s += wp[ci * 16] * ip[(size_t)ci * H * W];
    }
  }
  out[(((size_t)b * CoutTot + co) * 2 * H + oy) * Wo + ox] = s;
}

// ---------------------------------------------------------------------
// Elementwise / sampling helpers
// ---------------------------------------------------------------------
__device__ __forceinline__ void rs_idx(int n_in, int n_out, int o,
                                       int& i0, int& i1, float& w) {
  float s = (o + 0.5f) * ((float)n_in / (float)n_out) - 0.5f;
  s = fminf(fmaxf(s, 0.f), (float)(n_in - 1));
  float f = floorf(s);
  w = s - f; i0 = (int)f; i1 = min(i0 + 1, n_in - 1);
}

__device__ __forceinline__ float bil2(const float* src, int CT, int c,
                                      int Hi, int Wi, int b,
                                      int oy, int ox, int Ho, int Wo) {
  int y0, y1, x0, x1; float wy, wx;
  rs_idx(Hi, Ho, oy, y0, y1, wy);
  rs_idx(Wi, Wo, ox, x0, x1, wx);
  const float* p = src + (((size_t)b * CT + c) * Hi) * Wi;
  float a = p[(size_t)y0 * Wi + x0] * (1.f - wx) + p[(size_t)y0 * Wi + x1] * wx;
  float d = p[(size_t)y1 * Wi + x0] * (1.f - wx) + p[(size_t)y1 * Wi + x1] * wx;
  return a * (1.f - wy) + d * wy;
}

__global__ void k_resize(const float* __restrict__ src, int sCT, int s_off,
                         int Hi, int Wi,
                         float* __restrict__ dst, int dCT, int d_off,
                         int Ho, int Wo, int C, float mul,
                         const float* __restrict__ add, int aCT, int a_off)
{
  int ox = blockIdx.x * blockDim.x + threadIdx.x;
  int oy = blockIdx.y;
  int zc = blockIdx.z;
  int b = zc / C, c = zc - b * C;
  if (ox >= Wo) return;
  float v = mul * bil2(src, sCT, s_off + c, Hi, Wi, b, oy, ox, Ho, Wo);
  if (add) v += add[(((size_t)b * aCT + a_off + c) * Ho + oy) * Wo + ox];
  dst[(((size_t)b * dCT + d_off + c) * Ho + oy) * Wo + ox] = v;
}

__global__ void k_bwarp(const float* __restrict__ img, int CT, int c_off, int C,
                        int H, int W,
                        const float* __restrict__ flow, int fCT, int f_off,
                        float* __restrict__ out, int oCT, int o_off)
{
  int x = blockIdx.x * blockDim.x + threadIdx.x;
  int y = blockIdx.y, b = blockIdx.z;
  if (x >= W) return;
  size_t HW = (size_t)H * W;
  size_t fp = (((size_t)b * fCT + f_off) * H + y) * W + x;
  float fx = x + flow[fp];
  float fy = y + flow[fp + HW];
  fx = fminf(fmaxf(fx, 0.f), (float)(W - 1));
  fy = fminf(fmaxf(fy, 0.f), (float)(H - 1));
  float xf = floorf(fx), yf = floorf(fy);
  int x0 = (int)xf, y0 = (int)yf;
  float wx = fx - xf, wy = fy - yf;
  int x1 = min(x0 + 1, W - 1), y1 = min(y0 + 1, H - 1);
  for (int c = 0; c < C; ++c) {
    const float* ip = img + (((size_t)b * CT + c_off + c) * H) * W;
    float v00 = ip[(size_t)y0 * W + x0], v01 = ip[(size_t)y0 * W + x1];
    float v10 = ip[(size_t)y1 * W + x0], v11 = ip[(size_t)y1 * W + x1];
    float v = v00 * (1 - wx) * (1 - wy) + v01 * wx * (1 - wy)
            + v10 * (1 - wx) * wy       + v11 * wx * wy;
    out[(((size_t)b * oCT + o_off + c) * H + y) * W + x] = v;
  }
}

__global__ void k_copy_ch(const float* __restrict__ s, int sCT, int s_off,
                          float* __restrict__ d, int dCT, int d_off,
                          int C, int HW, int Bn)
{
  size_t n = (size_t)Bn * C * HW;
  for (size_t i = (size_t)blockIdx.x * blockDim.x + threadIdx.x; i < n;
       i += (size_t)gridDim.x * blockDim.x) {
    int b = (int)(i / ((size_t)C * HW));
    size_t r = i - (size_t)b * C * HW;
    int c = (int)(r / HW);
    size_t p = r - (size_t)c * HW;
    d[((size_t)b * dCT + d_off + c) * HW + p] =
        s[((size_t)b * sCT + s_off + c) * HW + p];
  }
}

__global__ void k_fill(float* __restrict__ p, float v, size_t n) {
  for (size_t i = (size_t)blockIdx.x * blockDim.x + threadIdx.x; i < n;
       i += (size_t)gridDim.x * blockDim.x) p[i] = v;
}

__global__ void k_bsum(const float* __restrict__ x0, const float* __restrict__ x1,
                       float* __restrict__ acc, int b, size_t n)
{
  __shared__ float red[256];
  float s = 0.f;
  for (size_t i = (size_t)blockIdx.x * blockDim.x + threadIdx.x; i < n;
       i += (size_t)gridDim.x * blockDim.x)
    s += x0[(size_t)b * n + i] + x1[(size_t)b * n + i];
  red[threadIdx.x] = s;
  __syncthreads();
  for (int o = 128; o > 0; o >>= 1) {
    if ((int)threadIdx.x < o) red[threadIdx.x] += red[threadIdx.x + o];
    __syncthreads();
  }
  if (threadIdx.x == 0) atomicAdd(acc + b, red[0]);
}

__global__ void k_norm(const float* __restrict__ x, const float* __restrict__ bsum,
                       float invN, float* __restrict__ out, size_t n3, int Bn)
{
  size_t n = (size_t)Bn * n3;
  for (size_t i = (size_t)blockIdx.x * blockDim.x + threadIdx.x; i < n;
       i += (size_t)gridDim.x * blockDim.x) {
    int b = (int)(i / n3);
    out[i] = x[i] * (1.f / 255.f) - bsum[b] * invN;
  }
}

__global__ void k_zmetric(const float* __restrict__ x, const float* __restrict__ bw,
                          int bCT, int b_off, float* __restrict__ z, int H, int W, int Bn)
{
  size_t HW = (size_t)H * W;
  size_t n = (size_t)Bn * HW;
  for (size_t i = (size_t)blockIdx.x * blockDim.x + threadIdx.x; i < n;
       i += (size_t)gridDim.x * blockDim.x) {
    int b = (int)(i / HW);
    size_t p = i - (size_t)b * HW;
    float s = 0.f;
    for (int c = 0; c < 3; ++c) {
      float a = x[((size_t)b * 3 + c) * HW + p];
      float w = bw[((size_t)b * bCT + b_off + c) * HW + p];
      s += fabsf(a - w);
    }
    float v = 1.f - s * (1.f / 3.f);
    v = fmaxf(v, 0.001f);
    z[i] = v * v;
  }
}

__global__ void k_sigz(const float* __restrict__ src, int sCT, int s_off,
                       float* __restrict__ dst, int H, int W, int Bn)
{
  size_t HW = (size_t)H * W;
  size_t n = (size_t)Bn * HW;
  for (size_t i = (size_t)blockIdx.x * blockDim.x + threadIdx.x; i < n;
       i += (size_t)gridDim.x * blockDim.x) {
    int b = (int)(i / HW);
    size_t p = i - (size_t)b * HW;
    float v = src[((size_t)b * sCT + s_off) * HW + p];
    dst[i] = (1.f / (1.f + __expf(-v))) * 0.99f + 0.01f;
  }
}

// Forward splat of 9 branches, one direction.  2x flow upsample fused.
__global__ void k_splat(const float* __restrict__ img3, const float* __restrict__ zfull,
                        const float* __restrict__ r, int rCT,
                        const float* __restrict__ fprev,
                        const float* __restrict__ tin, int oneMinus,
                        const float* __restrict__ alpha,
                        float* __restrict__ acc, int H, int W)
{
  int x = blockIdx.x * blockDim.x + threadIdx.x;
  int y = blockIdx.y, b = blockIdx.z;
  if (x >= W) return;
  size_t HW = (size_t)H * W;
  float t = tin[b];
  float td = oneMinus ? (1.f - t) : t;
  float m = zfull[(size_t)b * HW + (size_t)y * W + x] * alpha[0];
  m = fminf(fmaxf(m, -20.f), 20.f);
  float wg = td * __expf(m);
  const float* ip = img3 + (size_t)b * 3 * HW + (size_t)y * W + x;
  float p0 = ip[0] * wg, p1 = ip[HW] * wg, p2 = ip[2 * HW] * wg;
  float bfx = 2.f * bil2(fprev, 2, 0, H / 2, W / 2, b, y, x, H, W);
  float bfy = 2.f * bil2(fprev, 2, 1, H / 2, W / 2, b, y, x, H, W);
  float* ab = acc + (size_t)b * 4 * HW;
  for (int i = 0; i < 9; ++i) {
    float fx = x + (bfx + r[(((size_t)b * rCT + 2 * i) * H + y) * W + x]) * td;
    float fy = y + (bfy + r[(((size_t)b * rCT + 2 * i + 1) * H + y) * W + x]) * td;
    float xf = floorf(fx), yf = floorf(fy);
    int x0 = (int)xf, y0 = (int)yf;
    float ax = fx - xf, ay = fy - yf;
#pragma unroll
    for (int tap = 0; tap < 4; ++tap) {
      int xi = x0 + (tap & 1), yi = y0 + (tap >> 1);
      float wt = ((tap & 1) ? ax : 1.f - ax) * ((tap >> 1) ? ay : 1.f - ay);
      if (xi >= 0 && xi < W && yi >= 0 && yi < H && wt != 0.f) {
        size_t base = (size_t)yi * W + xi;
        atomicAdd(ab + 0 * HW + base, p0 * wt);
        atomicAdd(ab + 1 * HW + base, p1 * wt);
        atomicAdd(ab + 2 * HW + base, p2 * wt);
        atomicAdd(ab + 3 * HW + base, wg * wt);
      }
    }
  }
}

__global__ void k_final(const float* __restrict__ acc, const float* __restrict__ x0n,
                        const float* __restrict__ x1n, const float* __restrict__ tin,
                        const float* __restrict__ bsum, float invN,
                        float* __restrict__ out, int H, int W, int Bn)
{
  size_t HW = (size_t)H * W;
  size_t n = (size_t)Bn * HW;
  for (size_t i = (size_t)blockIdx.x * blockDim.x + threadIdx.x; i < n;
       i += (size_t)gridDim.x * blockDim.x) {
    int b = (int)(i / HW);
    size_t p = i - (size_t)b * HW;
    const float* ab = acc + (size_t)b * 4 * HW;
    float den = ab[3 * HW + p] + 18e-7f;
    float blank = (den < 1e-5f) ? 1.f : 0.f;
    float t = tin[b];
    float mean = bsum[b] * invN;
    float inv_den = 1.f / den;
    for (int c = 0; c < 3; ++c) {
      size_t ci = ((size_t)b * 3 + c) * HW + p;
      float pred = ab[(size_t)c * HW + p] * inv_den;
      float fb = ((1.f - t) * x0n[ci] + t * x1n[ci]) * blank;
      float v = fb + pred + mean;
      out[ci] = fminf(fmaxf(v, 0.f), 1.f);
    }
  }
}

// =====================================================================
// Host-side orchestration
// =====================================================================
namespace {

struct Arena {
  char* base; size_t off;
  float* a(size_t elems) {
    float* r = (float*)(base + off);
    off += ((elems * sizeof(float)) + 255) & ~(size_t)255;
    return r;
  }
};

struct CR { const float *a, *b, *w; };
struct Dec {
  const float *cr_a, *cr_b, *cr_w, *rb_a;
  const float *c1a, *c1b, *c1w, *c2a, *c2b, *c2w;
  const float *c3a, *c3b, *c3w, *c4a, *c4b, *c4w;
  const float *c5b, *c5w, *up_b, *up_w;
};

static Dec getDec(void* const* d, int i) {
  Dec p;
  p.cr_a=(const float*)d[i+0];  p.cr_b=(const float*)d[i+1];  p.cr_w=(const float*)d[i+2];
  p.rb_a=(const float*)d[i+3];
  p.c1a=(const float*)d[i+4];   p.c1b=(const float*)d[i+5];   p.c1w=(const float*)d[i+6];
  p.c2a=(const float*)d[i+7];   p.c2b=(const float*)d[i+8];   p.c2w=(const float*)d[i+9];
  p.c3a=(const float*)d[i+10];  p.c3b=(const float*)d[i+11];  p.c3w=(const float*)d[i+12];
  p.c4a=(const float*)d[i+13];  p.c4b=(const float*)d[i+14];  p.c4w=(const float*)d[i+15];
  p.c5b=(const float*)d[i+16];  p.c5w=(const float*)d[i+17];
  p.up_b=(const float*)d[i+18]; p.up_w=(const float*)d[i+19];
  return p;
}

static const int Bn = 2, FH = 512, FW = 896;

static void conv(hipStream_t st, const float* in, int CinT, int cin_off, int Cin,
                 int H, int W, int stride, float* out, int CoutT, int cout_off,
                 int Cout, const float* w, const float* b, const float* a,
                 const float* res, int resCT, int res_off) {
  int Ho = (stride == 2) ? H / 2 : H;
  int Wo = (stride == 2) ? W / 2 : W;
  dim3 g((Wo + 127) / 128, Bn * Ho, (Cout + 63) / 64);
  k_conv3x3_wmma<<<g, 256, 0, st>>>(in, CinT, cin_off, Cin, H, W,
                                    out, CoutT, cout_off, Cout, Ho, Wo, stride,
                                    w, b, a, res, resCT, res_off);
}

static void deconv(hipStream_t st, const float* in, int Cin, int H, int W,
                   float* out, int Cout, const float* w, const float* b) {
  dim3 g((2 * W + 63) / 64, 2 * H, Bn * Cout);
  k_deconv4x4s2<<<g, 64, 0, st>>>(in, Cin, H, W, out, Cout, Cout, w, b);
}

static void copy_ch(hipStream_t st, const float* s, int sCT, int s_off,
                    float* d, int dCT, int d_off, int C, int HW) {
  size_t n = (size_t)Bn * C * HW;
  int blk = (int)((n + 255) / 256); if (blk > 65535) blk = 65535;
  k_copy_ch<<<blk, 256, 0, st>>>(s, sCT, s_off, d, dCT, d_off, C, HW, Bn);
}

static void resize(hipStream_t st, const float* s, int sCT, int s_off,
                   int Hi, int Wi, float* d, int dCT, int d_off,
                   int Ho, int Wo, int C, float mul,
                   const float* add, int aCT, int a_off) {
  dim3 g((Wo + 255) / 256, Ho, Bn * C);
  k_resize<<<g, 256, 0, st>>>(s, sCT, s_off, Hi, Wi, d, dCT, d_off,
                              Ho, Wo, C, mul, add, aCT, a_off);
}

static void bwarp(hipStream_t st, const float* img, int CT, int c_off, int C,
                  int H, int W, const float* flow, int fCT, int f_off,
                  float* out, int oCT, int o_off) {
  dim3 g((W + 255) / 256, H, Bn);
  k_bwarp<<<g, 256, 0, st>>>(img, CT, c_off, C, H, W, flow, fCT, f_off,
                             out, oCT, o_off);
}

static void resblock(hipStream_t st, float* x, float* t1, float* t2, float* u32,
                     float* out, const Dec& P, int C, int H, int W) {
  int HW = H * W;
  conv(st, x,  C, 0, C, H, W, 1, t1, C, 0, C, P.c1w, P.c1b, P.c1a, nullptr, 0, 0);
  conv(st, t1, C, C - 32, 32, H, W, 1, u32, 32, 0, 32, P.c2w, P.c2b, P.c2a, nullptr, 0, 0);
  copy_ch(st, u32, 32, 0, t1, C, C - 32, 32, HW);
  conv(st, t1, C, 0, C, H, W, 1, t2, C, 0, C, P.c3w, P.c3b, P.c3a, nullptr, 0, 0);
  conv(st, t2, C, C - 32, 32, H, W, 1, u32, 32, 0, 32, P.c4w, P.c4b, P.c4a, nullptr, 0, 0);
  copy_ch(st, u32, 32, 0, t2, C, C - 32, 32, HW);
  conv(st, t2, C, 0, C, H, W, 1, out, C, 0, C, P.c5w, P.c5b, P.rb_a, x, C, 0);
}

static void dec_step(hipStream_t st, Arena& A, const Dec& P, int C, int mid,
                     const float* src, const float* tgt,
                     const float* flow, int fCT, int f_off, const float* z,
                     int H, int W, float* upout, int upC) {
  int HW = H * W;
  int Cin = 2 * C + 1;
  float* fin = A.a((size_t)Bn * Cin * HW);
  copy_ch(st, src, C, 0, fin, Cin, 0, C, HW);
  bwarp(st, tgt, C, 0, C, H, W, flow, fCT, f_off, fin, Cin, C);
  copy_ch(st, z, 1, 0, fin, Cin, 2 * C, 1, HW);
  float* m   = A.a((size_t)Bn * mid * HW);
  float* t1  = A.a((size_t)Bn * mid * HW);
  float* u32 = A.a((size_t)Bn * 32 * HW);
  conv(st, fin, Cin, 0, Cin, H, W, 1, m, mid, 0, mid,
       P.cr_w, P.cr_b, P.cr_a, nullptr, 0, 0);
  resblock(st, m, t1, /*t2=*/fin, u32, /*out=*/t1, P, mid, H, W);  // mid <= Cin
  deconv(st, t1, mid, H, W, upout, upC, P.up_w, P.up_b);
}

} // namespace

extern "C" void kernel_launch(void* const* d_in, const int* in_sizes, int n_in,
                              void* d_out, int out_size, void* d_ws, size_t ws_size,
                              hipStream_t stream) {
  if (n_in < 109) return;  // unexpected param flattening
  hipStream_t st = stream;

  const float* alpha = (const float*)d_in[0];
  Dec dec1 = getDec(d_in, 1),  dec2 = getDec(d_in, 21);
  Dec dec3 = getDec(d_in, 41), dec4 = getDec(d_in, 61);
  CR enc[8];
  for (int j = 0; j < 8; ++j)
    enc[j] = { (const float*)d_in[81 + 3 * j], (const float*)d_in[82 + 3 * j],
               (const float*)d_in[83 + 3 * j] };
  const float* x0  = (const float*)d_in[105];
  const float* x1  = (const float*)d_in[106];
  const float* tin = (const float*)d_in[108];

  const int HW = FH * FW;                 // 458752
  const int eC[4] = {32, 48, 72, 96};
  const int eH[4] = {256, 128, 64, 32};
  const int eW[4] = {448, 224, 112, 56};
  const int HW1 = 256 * 448, HW2 = 128 * 224, HW3 = 64 * 112, HW4 = 32 * 56;

  Arena A{ (char*)d_ws, 0 };

  // ---- persistent buffers ----
  float* bsum = A.a(Bn);
  float* x0n  = A.a((size_t)Bn * 3 * HW);
  float* x1n  = A.a((size_t)Bn * 3 * HW);
  float *f0[4], *f1[4];
  for (int l = 0; l < 4; ++l) f0[l] = A.a((size_t)Bn * eC[l] * eH[l] * eW[l]);
  for (int l = 0; l < 4; ++l) f1[l] = A.a((size_t)Bn * eC[l] * eH[l] * eW[l]);
  float* out4  = A.a((size_t)Bn * 4 * HW3);
  float* x0_4  = A.a((size_t)Bn * 3 * HW3);
  float* x1_4  = A.a((size_t)Bn * 3 * HW3);
  float* z0_4  = A.a((size_t)Bn * HW3);
  float* z1_4  = A.a((size_t)Bn * HW3);
  float* f01_3 = A.a((size_t)Bn * 2 * HW2);
  float* f10_3 = A.a((size_t)Bn * 2 * HW2);
  float* z0_3  = A.a((size_t)Bn * HW2);
  float* z1_3  = A.a((size_t)Bn * HW2);
  float* f01_2 = A.a((size_t)Bn * 2 * HW1);
  float* f10_2 = A.a((size_t)Bn * 2 * HW1);
  float* z0_2  = A.a((size_t)Bn * HW1);
  float* z1_2  = A.a((size_t)Bn * HW1);
  float* up1a  = A.a((size_t)Bn * 19 * HW);
  float* up1b  = A.a((size_t)Bn * 19 * HW);
  float* z0_1  = A.a((size_t)Bn * HW);
  float* z1_1  = A.a((size_t)Bn * HW);
  float* accum = A.a((size_t)Bn * 4 * HW);
  const size_t mark0 = A.off;

  // ---- preprocess: mean-sub normalize ----
  k_fill<<<1, 256, 0, st>>>(bsum, 0.f, Bn);
  const size_t n3 = (size_t)3 * HW;
  k_bsum<<<1024, 256, 0, st>>>(x0, x1, bsum, 0, n3);
  k_bsum<<<1024, 256, 0, st>>>(x0, x1, bsum, 1, n3);
  const float invN = 1.f / (255.f * (float)(3 * 2 * FH * FW));
  k_norm<<<4096, 256, 0, st>>>(x0, bsum, invN, x0n, n3, Bn);
  k_norm<<<4096, 256, 0, st>>>(x1, bsum, invN, x1n, n3, Bn);

  // ---- encoder (x2 images) ----
  {
    float* tmpE = A.a((size_t)Bn * 32 * HW1);  // largest stage output
    for (int im = 0; im < 2; ++im) {
      const float* cur = im ? x1n : x0n;
      int cC = 3, cH = FH, cW = FW;
      float** fs = im ? f1 : f0;
      for (int l = 0; l < 4; ++l) {
        conv(st, cur, cC, 0, cC, cH, cW, 2, tmpE, eC[l], 0, eC[l],
             enc[2 * l].w, enc[2 * l].b, enc[2 * l].a, nullptr, 0, 0);
        conv(st, tmpE, eC[l], 0, eC[l], eH[l], eW[l], 1, fs[l], eC[l], 0, eC[l],
             enc[2 * l + 1].w, enc[2 * l + 1].b, enc[2 * l + 1].a, nullptr, 0, 0);
        cur = fs[l]; cC = eC[l]; cH = eH[l]; cW = eW[l];
      }
    }
    A.off = mark0;
  }

  // ---- dec4: coarsest flow ----
  {
    float* cat = A.a((size_t)Bn * 192 * HW4);
    copy_ch(st, f0[3], 96, 0, cat, 192, 0,  96, HW4);
    copy_ch(st, f1[3], 96, 0, cat, 192, 96, 96, HW4);
    float* m4  = A.a((size_t)Bn * 192 * HW4);
    float* t1  = A.a((size_t)Bn * 192 * HW4);
    float* u32 = A.a((size_t)Bn * 32 * HW4);
    conv(st, cat, 192, 0, 192, 32, 56, 1, m4, 192, 0, 192,
         dec4.cr_w, dec4.cr_b, dec4.cr_a, nullptr, 0, 0);
    resblock(st, m4, t1, /*t2=*/cat, u32, /*out=*/t1, dec4, 192, 32, 56);
    deconv(st, t1, 192, 32, 56, out4, 4, dec4.up_w, dec4.up_b);
    A.off = mark0;
  }

  // ---- level-4 photometric confidence ----
  resize(st, x0n, 3, 0, FH, FW, x0_4, 3, 0, 64, 112, 3, 1.f, nullptr, 0, 0);
  resize(st, x1n, 3, 0, FH, FW, x1_4, 3, 0, 64, 112, 3, 1.f, nullptr, 0, 0);
  {
    float* bw = A.a((size_t)Bn * 3 * HW3);
    bwarp(st, x1_4, 3, 0, 3, 64, 112, out4, 4, 0, bw, 3, 0);
    k_zmetric<<<512, 256, 0, st>>>(x0_4, bw, 3, 0, z0_4, 64, 112, Bn);
    bwarp(st, x0_4, 3, 0, 3, 64, 112, out4, 4, 2, bw, 3, 0);
    k_zmetric<<<512, 256, 0, st>>>(x1_4, bw, 3, 0, z1_4, 64, 112, Bn);
    A.off = mark0;
  }

  // ---- dec3 ----
  {
    float* up = A.a((size_t)Bn * 3 * HW2);
    dec_step(st, A, dec3, 72, 144, f0[2], f1[2], out4, 4, 0, z0_4, 64, 112, up, 3);
    resize(st, out4, 4, 0, 64, 112, f01_3, 2, 0, 128, 224, 2, 2.f, up, 3, 0);
    k_sigz<<<512, 256, 0, st>>>(up, 3, 2, z0_3, 128, 224, Bn);
    resize(st, z0_4, 1, 0, 64, 112, z0_3, 1, 0, 128, 224, 1, 1.f, z0_3, 1, 0);
    A.off = mark0;
    up = A.a((size_t)Bn * 3 * HW2);
    dec_step(st, A, dec3, 72, 144, f1[2], f0[2], out4, 4, 2, z1_4, 64, 112, up, 3);
    resize(st, out4, 4, 2, 64, 112, f10_3, 2, 0, 128, 224, 2, 2.f, up, 3, 0);
    k_sigz<<<512, 256, 0, st>>>(up, 3, 2, z1_3, 128, 224, Bn);
    resize(st, z1_4, 1, 0, 64, 112, z1_3, 1, 0, 128, 224, 1, 1.f, z1_3, 1, 0);
    A.off = mark0;
  }

  // ---- dec2 ----
  {
    float* up = A.a((size_t)Bn * 3 * HW1);
    dec_step(st, A, dec2, 48, 96, f0[1], f1[1], f01_3, 2, 0, z0_3, 128, 224, up, 3);
    resize(st, f01_3, 2, 0, 128, 224, f01_2, 2, 0, 256, 448, 2, 2.f, up, 3, 0);
    k_sigz<<<1024, 256, 0, st>>>(up, 3, 2, z0_2, 256, 448, Bn);
    resize(st, z0_3, 1, 0, 128, 224, z0_2, 1, 0, 256, 448, 1, 1.f, z0_2, 1, 0);
    A.off = mark0;
    up = A.a((size_t)Bn * 3 * HW1);
    dec_step(st, A, dec2, 48, 96, f1[1], f0[1], f10_3, 2, 0, z1_3, 128, 224, up, 3);
    resize(st, f10_3, 2, 0, 128, 224, f10_2, 2, 0, 256, 448, 2, 2.f, up, 3, 0);
    k_sigz<<<1024, 256, 0, st>>>(up, 3, 2, z1_2, 256, 448, Bn);
    resize(st, z1_3, 1, 0, 128, 224, z1_2, 1, 0, 256, 448, 1, 1.f, z1_2, 1, 0);
    A.off = mark0;
  }

  // ---- dec1 (9-branch multi-flow) ----
  dec_step(st, A, dec1, 32, 64, f0[0], f1[0], f01_2, 2, 0, z0_2, 256, 448, up1a, 19);
  A.off = mark0;
  dec_step(st, A, dec1, 32, 64, f1[0], f0[0], f10_2, 2, 0, z1_2, 256, 448, up1b, 19);
  A.off = mark0;
  k_sigz<<<4096, 256, 0, st>>>(up1a, 19, 18, z0_1, FH, FW, Bn);
  resize(st, z0_2, 1, 0, 256, 448, z0_1, 1, 0, FH, FW, 1, 1.f, z0_1, 1, 0);
  k_sigz<<<4096, 256, 0, st>>>(up1b, 19, 18, z1_1, FH, FW, Bn);
  resize(st, z1_2, 1, 0, 256, 448, z1_1, 1, 0, FH, FW, 1, 1.f, z1_1, 1, 0);

  // ---- forward splat (both directions, 9 branches each) ----
  k_fill<<<4096, 256, 0, st>>>(accum, 0.f, (size_t)Bn * 4 * HW);
  {
    dim3 g((FW + 255) / 256, FH, Bn);
    k_splat<<<g, 256, 0, st>>>(x0n, z0_1, up1a, 19, f01_2, tin, 0, alpha, accum, FH, FW);
    k_splat<<<g, 256, 0, st>>>(x1n, z1_1, up1b, 19, f10_2, tin, 1, alpha, accum, FH, FW);
  }

  // ---- combine ----
  k_final<<<4096, 256, 0, st>>>(accum, x0n, x1n, tin, bsum, invN,
                                (float*)d_out, FH, FW, Bn);
}
